// CNFVectorModule_78829829751257
// MI455X (gfx1250) — matrix operand order repaired and checked by reading, not compiled, not run on hardware
//
#include <hip/hip_runtime.h>
#include <hip/hip_bf16.h>
#include <math.h>

// ---------------------------------------------------------------------------
// CNF vector module for MI455X (gfx1250, wave32).
// Dominant cost: 6 per-pair filter MLPs (RBF32 -> 32 swish -> 16) over 1M
// pairs ~ 19 GFLOP. RBF tile computed on the fly in LDS; filter GEMMs via
// v_wmma_f32_16x16x32_f16 (K=32 matches RBF and hidden widths).
// Swish uses v_tanh_f32 (1 TRANS) when available, else v_rcp_f32 path.
// Parallelism: each i's j-reduction is split across JS=4 waves (4096 waves
// total, ~4/SIMD) so TRANS stalls and WMMA latency are hidden by wave
// interleaving; partial sums combined through LDS.
// ---------------------------------------------------------------------------

typedef __attribute__((ext_vector_type(16))) _Float16 v16h;
typedef __attribute__((ext_vector_type(8)))  float    v8f;

#define NATOMS 1024
#define NC     16     // C0 per filter output
#define NRBF   32
#define NFH    32
#define JTILE  16
#define WPB_I  2      // atoms (i) per block
#define JS     4      // j-range splits per atom
#define NWAVES (WPB_I * JS)
#define R_CUT_F 5.0f
#define EPS_F   1e-7f
#define PI_F    3.14159265358979f

__device__ __forceinline__ float rcp_f(float x) { return __builtin_amdgcn_rcpf(x); }

#if defined(__gfx1250__) && __has_builtin(__builtin_amdgcn_tanhf)
// swish(x) = x*sigmoid(x) = h + h*tanh(h), h = x/2  -> single v_tanh_f32
__device__ __forceinline__ float swish_f(float x) {
  const float h = 0.5f * x;
  return h + h * __builtin_amdgcn_tanhf(h);
}
__device__ __forceinline__ float sigm_f(float x) {
  return 0.5f + 0.5f * __builtin_amdgcn_tanhf(0.5f * x);
}
#else
__device__ __forceinline__ float swish_f(float x) { return x * rcp_f(1.0f + __expf(-x)); }
__device__ __forceinline__ float sigm_f(float x)  { return rcp_f(1.0f + __expf(-x)); }
#endif

// Compiler-only memory fence: same-wave DS ops complete in order on CDNA5,
// we just need to stop IR-level reordering of the LDS store->load repack.
__device__ __forceinline__ void lds_compiler_fence() { __asm__ volatile("" ::: "memory"); }

// ---- WMMA operand loaders --------------------------------------------------
// A-matrix (16x32, f16) VGPR layout (CDNA5 ISA 7.12.2):
//   lane&15 = row M (both halves); half = lane>>4 selects K subset:
//   half 0: e<8 -> K=e,   e>=8 -> K=e+8    ({0..7,16..23})
//   half 1: e<8 -> K=e+8, e>=8 -> K=e+16   ({8..15,24..31})
__device__ __forceinline__ int a_kmap(int e, int half) {
  return (e < 8) ? (e + 8 * half) : (e + 8 + 8 * half);
}

__device__ __forceinline__ v16h load_A_lds(const _Float16* t /*16x32 row-major*/, int lane) {
  const int m = lane & 15, half = lane >> 4;
  v16h a;
#pragma unroll
  for (int e = 0; e < 16; ++e) a[e] = t[m * NRBF + a_kmap(e, half)];
  return a;
}

// B-matrix (32x16, f16): lane&15 = col N; half = lane>>4; element e -> K = a_kmap(e, half)
// (same K order as the A operand). W is f32 row-major [K][ncols].
__device__ __forceinline__ v16h load_B_glb(const float* __restrict__ W, int ncols,
                                           int colBase, int lane) {
  const int n = lane & 15, half = lane >> 4;
  v16h b;
#pragma unroll
  for (int e = 0; e < 16; ++e)
    b[e] = (_Float16)W[a_kmap(e, half) * ncols + colBase + n];
  return b;
}

struct FilterW { v16h B0a, B0b, B1; float b0a, b0b, b1; };

// Filter MLP on a 16-pair tile: R = (swish(rbf@W0 + b0)) @ W1 + b1
// Result in C/D layout: lane holds column c=lane&15, VGPR v holds row v+8*(lane>>4).
__device__ __forceinline__ v8f apply_filter(const _Float16* rbf, _Float16* hbuf,
                                            const FilterW& f, int lane) {
  v8f z = {};
  v16h a = load_A_lds(rbf, lane);
  v8f h0 = __builtin_amdgcn_wmma_f32_16x16x32_f16(false, a, false, f.B0a, (short)0, z, false, false);
  v8f h1 = __builtin_amdgcn_wmma_f32_16x16x32_f16(false, a, false, f.B0b, (short)0, z, false, false);
  const int c = lane & 15, half = lane >> 4;
#pragma unroll
  for (int v = 0; v < 8; ++v) {
    const int row = v + 8 * half;
    hbuf[row * NFH + c]      = (_Float16)swish_f(h0[v] + f.b0a);
    hbuf[row * NFH + 16 + c] = (_Float16)swish_f(h1[v] + f.b0b);
  }
  lds_compiler_fence();              // same-wave DS is in-order; no barrier needed
  v16h ah = load_A_lds(hbuf, lane);
  v8f r = __builtin_amdgcn_wmma_f32_16x16x32_f16(false, ah, false, f.B1, (short)0, z, false, false);
#pragma unroll
  for (int v = 0; v < 8; ++v) r[v] += f.b1;
  return r;
}

// ---- pair interaction layer ------------------------------------------------
struct LayerArgs {
  const float* pos;     // N x 3
  const float* L0;      // N x 16
  const float* L1;      // N x 16 x 3 (layer 2 only)
  const float* W0[4];   // filters in order [f00, f01, f10, f11]: 32x32
  const float* W1[4];   //                                       32x16
  const float* b0[4];   //                                       32
  const float* b1[4];   //                                       16
  float* out0;          // N x 16
  float* out1;          // N x 16 x 3
  float* p10;           // N x 16      (layer 2)
  float* p11;           // N x 16 x 3  (layer 2)
};

template <int NF>
__global__ __launch_bounds__(NWAVES * 32)
void pair_layer_kernel(LayerArgs A) {
  // per-wave buffers (depend on i and j-split)
  __shared__ _Float16 s_rbf[NWAVES][JTILE * NRBF];
  __shared__ _Float16 s_h  [NWAVES][JTILE * NFH];
  __shared__ float    s_nu [NWAVES][JTILE * 4];      // norm, ux, uy, uz per j
  // per-j-split shared node features (i-independent within a split group)
  __shared__ float    s_L0 [JS][JTILE * NC];
  __shared__ float    s_L1 [JS][JTILE * NC * 3];
  // cross-split partial-sum combine
  __shared__ float    s_red[WPB_I][JS][NC][8];

  const int tid  = threadIdx.x;
  const int wid  = tid >> 5;
  const int lane = tid & 31;
  const int iw   = wid % WPB_I;                      // which i in the block
  const int js   = wid / WPB_I;                      // which j-split
  const int i    = blockIdx.x * WPB_I + iw;

  const float pix = A.pos[3 * i + 0];
  const float piy = A.pos[3 * i + 1];
  const float piz = A.pos[3 * i + 2];

  FilterW fw[NF];
#pragma unroll
  for (int f = 0; f < NF; ++f) {
    fw[f].B0a = load_B_glb(A.W0[f], NFH, 0, lane);
    fw[f].B0b = load_B_glb(A.W0[f], NFH, 16, lane);
    fw[f].B1  = load_B_glb(A.W1[f], NC, 0, lane);
    fw[f].b0a = A.b0[f][lane & 15];
    fw[f].b0b = A.b0[f][16 + (lane & 15)];
    fw[f].b1  = A.b1[f][lane & 15];
  }

  const int c = lane & 15, half = lane >> 4;
  float acc0 = 0.f, acc1x = 0.f, acc1y = 0.f, acc1z = 0.f;
  float accP0 = 0.f, accP1x = 0.f, accP1y = 0.f, accP1z = 0.f;

  const int jbeg = js * (NATOMS / JS);
  const int jend = jbeg + (NATOMS / JS);
  for (int j0 = jbeg; j0 < jend; j0 += JTILE) {      // uniform 16 tiles per wave
    // per-wave: distances + unit vectors for this i
    if (lane < JTILE) {
      const int jg = j0 + lane;
      const float dx = pix - A.pos[3 * jg + 0];
      const float dy = piy - A.pos[3 * jg + 1];
      const float dz = piz - A.pos[3 * jg + 2];
      const float nrm = sqrtf(dx * dx + dy * dy + dz * dz);
      const float inv = rcp_f(nrm + EPS_F);
      s_nu[wid][lane * 4 + 0] = (jg == i) ? 0.0f : nrm;   // masked norm
      s_nu[wid][lane * 4 + 1] = dx * inv;
      s_nu[wid][lane * 4 + 2] = dy * inv;
      s_nu[wid][lane * 4 + 3] = dz * inv;
    }
    // split-group cooperative: stage L0 (and L1) j-tile once per group
    const int gtid = iw * 32 + lane;                 // 0 .. WPB_I*32-1
    for (int e = gtid; e < JTILE * NC; e += WPB_I * 32)
      s_L0[js][e] = A.L0[j0 * NC + e];
    if (NF == 4) {
      for (int e = gtid; e < JTILE * NC * 3; e += WPB_I * 32)
        s_L1[js][e] = A.L1[j0 * NC * 3 + e];
    }
    __syncthreads();   // staged tiles visible within each split group

    // RBF tile: lane = frequency index k (0..31); phase-split sin/rcp chains.
    {
      const float cnorm = 0.632455532f;                       // sqrt(2/5)
      const float w = (float)(lane + 1) * (PI_F / R_CUT_F);
      float nr[JTILE], sn[JTILE], iv[JTILE];
#pragma unroll
      for (int r = 0; r < JTILE; ++r) nr[r] = s_nu[wid][r * 4];
#pragma unroll
      for (int r = 0; r < JTILE; ++r) sn[r] = __sinf(w * nr[r]);
#pragma unroll
      for (int r = 0; r < JTILE; ++r) iv[r] = rcp_f(nr[r] + EPS_F);
#pragma unroll
      for (int r = 0; r < JTILE; ++r) {
        const float v = (nr[r] > 0.f) ? cnorm * sn[r] * iv[r] : 0.f;
        s_rbf[wid][r * NRBF + lane] = (_Float16)v;
      }
    }
    lds_compiler_fence();            // same-wave rbf store -> load ordering

    // f00 -> out0[i,c] = sum_j R00[i,j,c] * L0[j,c]
    {
      v8f r = apply_filter(s_rbf[wid], s_h[wid], fw[0], lane);
#pragma unroll
      for (int v = 0; v < 8; ++v) {
        const int rl = v + 8 * half;
        const float mk = (j0 + rl == i) ? 0.f : 1.f;
        acc0 += r[v] * mk * s_L0[js][rl * NC + c];
      }
    }
    // f01 -> out1[i,c,m] = sum_j R01[i,j,c] * unit[i,j,m] * L0[j,c]
    {
      v8f r = apply_filter(s_rbf[wid], s_h[wid], fw[1], lane);
#pragma unroll
      for (int v = 0; v < 8; ++v) {
        const int rl = v + 8 * half;
        const float mk = (j0 + rl == i) ? 0.f : 1.f;
        const float wv = r[v] * mk * s_L0[js][rl * NC + c];
        acc1x += wv * s_nu[wid][rl * 4 + 1];
        acc1y += wv * s_nu[wid][rl * 4 + 2];
        acc1z += wv * s_nu[wid][rl * 4 + 3];
      }
    }
    if (NF == 4) {
      // f10 -> p10[i,c] = sum_j R10[i,j,c] * dot(unit[i,j,:], L1[j,c,:])
      {
        v8f r = apply_filter(s_rbf[wid], s_h[wid], fw[2], lane);
#pragma unroll
        for (int v = 0; v < 8; ++v) {
          const int rl = v + 8 * half;
          const float mk = (j0 + rl == i) ? 0.f : 1.f;
          const float* l1 = &s_L1[js][(rl * NC + c) * 3];
          const float d = s_nu[wid][rl * 4 + 1] * l1[0] +
                          s_nu[wid][rl * 4 + 2] * l1[1] +
                          s_nu[wid][rl * 4 + 3] * l1[2];
          accP0 += r[v] * mk * d;
        }
      }
      // f11 -> p11[i,c,m] = sum_j R11[i,j,c] * L1[j,c,m]
      {
        v8f r = apply_filter(s_rbf[wid], s_h[wid], fw[3], lane);
#pragma unroll
        for (int v = 0; v < 8; ++v) {
          const int rl = v + 8 * half;
          const float mk = (j0 + rl == i) ? 0.f : 1.f;
          const float* l1 = &s_L1[js][(rl * NC + c) * 3];
          const float wv = r[v] * mk;
          accP1x += wv * l1[0];
          accP1y += wv * l1[1];
          accP1z += wv * l1[2];
        }
      }
    }
    __syncthreads();   // group done with s_L0/s_L1 before next tile restages
  }

  // fold lane halves (rows 0..7 in lanes<16, rows 8..15 in lanes>=16)
  acc0  += __shfl_xor(acc0, 16);
  acc1x += __shfl_xor(acc1x, 16);
  acc1y += __shfl_xor(acc1y, 16);
  acc1z += __shfl_xor(acc1z, 16);
  if (NF == 4) {
    accP0  += __shfl_xor(accP0, 16);
    accP1x += __shfl_xor(accP1x, 16);
    accP1y += __shfl_xor(accP1y, 16);
    accP1z += __shfl_xor(accP1z, 16);
  }
  // combine the JS partial sums per i through LDS
  if (lane < 16) {
    float* r = s_red[iw][js][lane];
    r[0] = acc0; r[1] = acc1x; r[2] = acc1y; r[3] = acc1z;
    r[4] = accP0; r[5] = accP1x; r[6] = accP1y; r[7] = accP1z;
  }
  __syncthreads();
  if (js == 0 && lane < 16) {
    float t[8] = {0.f, 0.f, 0.f, 0.f, 0.f, 0.f, 0.f, 0.f};
#pragma unroll
    for (int s = 0; s < JS; ++s)
#pragma unroll
      for (int k = 0; k < 8; ++k) t[k] += s_red[iw][s][lane][k];
    A.out0[i * NC + c] = t[0];
    A.out1[(i * NC + c) * 3 + 0] = t[1];
    A.out1[(i * NC + c) * 3 + 1] = t[2];
    A.out1[(i * NC + c) * 3 + 2] = t[3];
    if (NF == 4) {
      A.p10[i * NC + c] = t[4];
      A.p11[(i * NC + c) * 3 + 0] = t[5];
      A.p11[(i * NC + c) * 3 + 1] = t[6];
      A.p11[(i * NC + c) * 3 + 2] = t[7];
    }
  }
}

// ---- small kernels ---------------------------------------------------------
__global__ void time_mlp_kernel(const float* __restrict__ t,
                                const float* __restrict__ W0, const float* __restrict__ W1,
                                const float* __restrict__ b0, const float* __restrict__ b1,
                                float* __restrict__ emb) {
  __shared__ float h[64];
  const int tid = threadIdx.x;
  if (tid < 64) h[tid] = swish_f(t[0] * W0[tid] + b0[tid]);
  __syncthreads();
  if (tid < 8) {
    float s = b1[tid];
    for (int k = 0; k < 64; ++k) s += h[k] * W1[k * 8 + tid];
    emb[tid] = s;
  }
}

__global__ void build_L0_kernel(const float* __restrict__ f0, const float* __restrict__ emb,
                                float* __restrict__ L0) {
  const int idx = blockIdx.x * blockDim.x + threadIdx.x;
  if (idx >= NATOMS * NC) return;
  const int n = idx >> 4, ch = idx & 15;
  L0[idx] = (ch < 8) ? f0[n * 8 + ch] : emb[ch - 8];
}

// Layer-1 per-node update: L0' = mlp0(out0); L1 = (out1@mix1) * sigmoid(gate(|.|))
__global__ void node_update1_kernel(const float* __restrict__ out0, const float* __restrict__ out1,
                                    const float* __restrict__ W0, const float* __restrict__ W1,
                                    const float* __restrict__ b0, const float* __restrict__ b1,
                                    const float* __restrict__ mix,
                                    const float* __restrict__ gW0, const float* __restrict__ gW1,
                                    const float* __restrict__ gb0, const float* __restrict__ gb1,
                                    float* __restrict__ L0n, float* __restrict__ L1n) {
  const int n = blockIdx.x * blockDim.x + threadIdx.x;
  if (n >= NATOMS) return;
  float x[NC];
  for (int ci = 0; ci < NC; ++ci) x[ci] = out0[n * NC + ci];
  float h[32];
  for (int k = 0; k < 32; ++k) {
    float s = b0[k];
    for (int ci = 0; ci < NC; ++ci) s += x[ci] * W0[ci * 32 + k];
    h[k] = swish_f(s);
  }
  for (int ci = 0; ci < NC; ++ci) {
    float s = b1[ci];
    for (int k = 0; k < 32; ++k) s += h[k] * W1[k * NC + ci];
    L0n[n * NC + ci] = s;
  }
  float vm[NC][3], vn[NC];
  for (int d = 0; d < NC; ++d) {
    float m0 = 0.f, m1 = 0.f, m2 = 0.f;
    for (int ci = 0; ci < NC; ++ci) {
      const float w = mix[ci * NC + d];
      m0 += out1[(n * NC + ci) * 3 + 0] * w;
      m1 += out1[(n * NC + ci) * 3 + 1] * w;
      m2 += out1[(n * NC + ci) * 3 + 2] * w;
    }
    vm[d][0] = m0; vm[d][1] = m1; vm[d][2] = m2;
    vn[d] = sqrtf(m0 * m0 + m1 * m1 + m2 * m2 + EPS_F);
  }
  float g[32];
  for (int k = 0; k < 32; ++k) {
    float s = gb0[k];
    for (int d = 0; d < NC; ++d) s += vn[d] * gW0[d * 32 + k];
    g[k] = swish_f(s);
  }
  for (int d = 0; d < NC; ++d) {
    float s = gb1[d];
    for (int k = 0; k < 32; ++k) s += g[k] * gW1[k * NC + d];
    const float gate = sigm_f(s);
    L1n[(n * NC + d) * 3 + 0] = vm[d][0] * gate;
    L1n[(n * NC + d) * 3 + 1] = vm[d][1] * gate;
    L1n[(n * NC + d) * 3 + 2] = vm[d][2] * gate;
  }
}

// Layer-2 per-node update + final output assembly.
__global__ void node_update2_kernel(const float* __restrict__ out0, const float* __restrict__ p10,
                                    const float* __restrict__ out1, const float* __restrict__ p11,
                                    const float* __restrict__ W0, const float* __restrict__ W1,
                                    const float* __restrict__ b0, const float* __restrict__ b1,
                                    const float* __restrict__ mix,
                                    const float* __restrict__ gW0, const float* __restrict__ gW1,
                                    const float* __restrict__ gb0, const float* __restrict__ gb1,
                                    const float* __restrict__ y, float* __restrict__ out) {
  const int n = blockIdx.x * blockDim.x + threadIdx.x;
  if (n >= NATOMS) return;
  float x[32];
  for (int ci = 0; ci < 16; ++ci) { x[ci] = out0[n * 16 + ci]; x[16 + ci] = p10[n * 16 + ci]; }
  float h[32];
  for (int k = 0; k < 32; ++k) {
    float s = b0[k];
    for (int ci = 0; ci < 32; ++ci) s += x[ci] * W0[ci * 32 + k];
    h[k] = swish_f(s);
  }
  float s0 = b1[0];
  for (int k = 0; k < 32; ++k) s0 += h[k] * W1[k];         // W1 is 32x1
  float vm[2][3] = {{0.f, 0.f, 0.f}, {0.f, 0.f, 0.f}};
  for (int ci = 0; ci < 32; ++ci) {
    const float w0 = mix[ci * 2 + 0], w1 = mix[ci * 2 + 1];
    for (int m = 0; m < 3; ++m) {
      const float v = (ci < 16) ? out1[(n * 16 + ci) * 3 + m]
                                : p11[(n * 16 + (ci - 16)) * 3 + m];
      vm[0][m] += v * w0;
      vm[1][m] += v * w1;
    }
  }
  const float vn0 = sqrtf(vm[0][0] * vm[0][0] + vm[0][1] * vm[0][1] + vm[0][2] * vm[0][2] + EPS_F);
  const float vn1 = sqrtf(vm[1][0] * vm[1][0] + vm[1][1] * vm[1][1] + vm[1][2] * vm[1][2] + EPS_F);
  float g[16];
  for (int k = 0; k < 16; ++k)
    g[k] = swish_f(gb0[k] + vn0 * gW0[k] + vn1 * gW0[16 + k]);  // gW0 is 2x16
  float gate1v = gb1[1];
  for (int k = 0; k < 16; ++k) gate1v += g[k] * gW1[k * 2 + 1]; // only d=1 needed
  gate1v = sigm_f(gate1v);
  const float* vel = y + 3 * NATOMS;                            // y[1]
  for (int m = 0; m < 3; ++m) {
    const float v  = vel[n * 3 + m];
    const float tr = vm[1][m] * gate1v;                         // translations
    out[n * 3 + m]              = v;                            // ydot[0]
    out[3 * NATOMS + n * 3 + m] = s0 * v + tr;                  // ydot[1]
    out[6 * NATOMS + n * 3 + m] = s0;                           // scales (repeat x3)
    out[9 * NATOMS + n * 3 + m] = tr;                           // translations
  }
}

// ---- input index map (flattened pytree, dict insertion order) --------------
// 0:t 1:y 2:f0
// 3..6:   time   {W0(1x64), W1(64x8), b0(64), b1(8)}
// layer0: 7..10  f00{W0,W1,b0,b1}   11..14 f01{...}
//         15..18 mlp0{W0(16x32),W1(32x16),b0,b1}  19 mix1(16x16)
//         20..23 gate1{W0(16x32),W1(32x16),b0,b1}
// layer1: 24..27 f00  28..31 f10  32..35 f01  36..39 f11
//         40..43 mlp0{W0(32x32),W1(32x1),b0,b1}  44 mix1(32x2)
//         45..48 gate1{W0(2x16),W1(16x2),b0,b1}

extern "C" void kernel_launch(void* const* d_in, const int* in_sizes, int n_in,
                              void* d_out, int out_size, void* d_ws, size_t ws_size,
                              hipStream_t stream) {
  (void)in_sizes; (void)n_in; (void)out_size; (void)ws_size;
  auto in = [&](int k) { return (const float*)d_in[k]; };

  float* ws    = (float*)d_ws;
  float* emb   = ws;                               // 8
  float* L0a   = emb + 8;                          // N*16
  float* out0a = L0a + NATOMS * NC;                // N*16
  float* out1a = out0a + NATOMS * NC;              // N*48
  float* L0b   = out1a + NATOMS * NC * 3;          // N*16
  float* L1b   = L0b + NATOMS * NC;                // N*48
  float* out0b = L1b + NATOMS * NC * 3;            // N*16
  float* p10b  = out0b + NATOMS * NC;              // N*16
  float* out1b = p10b + NATOMS * NC;               // N*48
  float* p11b  = out1b + NATOMS * NC * 3;          // N*48

  // time embedding + initial node features L0 = [f0 | t_emb]
  time_mlp_kernel<<<1, 64, 0, stream>>>(in(0), in(3), in(4), in(5), in(6), emb);
  build_L0_kernel<<<(NATOMS * NC + 255) / 256, 256, 0, stream>>>(in(2), emb, L0a);

  // layer 1: two filters
  LayerArgs a1 = {};
  a1.pos = in(1);  a1.L0 = L0a;  a1.L1 = nullptr;
  a1.W0[0] = in(7);  a1.W1[0] = in(8);  a1.b0[0] = in(9);  a1.b1[0] = in(10);   // f00
  a1.W0[1] = in(11); a1.W1[1] = in(12); a1.b0[1] = in(13); a1.b1[1] = in(14);   // f01
  a1.out0 = out0a; a1.out1 = out1a; a1.p10 = nullptr; a1.p11 = nullptr;
  pair_layer_kernel<2><<<NATOMS / WPB_I, NWAVES * 32, 0, stream>>>(a1);

  node_update1_kernel<<<(NATOMS + 127) / 128, 128, 0, stream>>>(
      out0a, out1a, in(15), in(16), in(17), in(18), in(19),
      in(20), in(21), in(22), in(23), L0b, L1b);

  // layer 2: four filters (order in args: f00, f01, f10, f11)
  LayerArgs a2 = {};
  a2.pos = in(1);  a2.L0 = L0b;  a2.L1 = L1b;
  a2.W0[0] = in(24); a2.W1[0] = in(25); a2.b0[0] = in(26); a2.b1[0] = in(27);   // f00
  a2.W0[1] = in(32); a2.W1[1] = in(33); a2.b0[1] = in(34); a2.b1[1] = in(35);   // f01
  a2.W0[2] = in(28); a2.W1[2] = in(29); a2.b0[2] = in(30); a2.b1[2] = in(31);   // f10
  a2.W0[3] = in(36); a2.W1[3] = in(37); a2.b0[3] = in(38); a2.b1[3] = in(39);   // f11
  a2.out0 = out0b; a2.out1 = out1b; a2.p10 = p10b; a2.p11 = p11b;
  pair_layer_kernel<4><<<NATOMS / WPB_I, NWAVES * 32, 0, stream>>>(a2);

  node_update2_kernel<<<(NATOMS + 127) / 128, 128, 0, stream>>>(
      out0b, p10b, out1b, p11b, in(40), in(41), in(42), in(43), in(44),
      in(45), in(46), in(47), in(48), in(1), (float*)d_out);
}
